// context_prediction_38972533243992
// MI455X (gfx1250) — 8-bit matrix operand order repaired and checked by reading, compiled, not run on hardware
//
#include <hip/hip_runtime.h>

#define B_    8
#define CIN   192
#define COUT  384
#define H_    64
#define W_    64
#define HP    68
#define WP    68
#define KW    5
#define NTAPS 12
#define KGEMM (NTAPS * CIN)
#define NITER (NTAPS * 3)

typedef int v8i __attribute__((ext_vector_type(8)));

#define WS_F_OFF  64
#define WS_BQ_OFF 3200
#define WS_WQ_OFF 8192
#define WS_XQ_OFF (1u << 20)
#define XQ_BYTES  (B_ * HP * WP * CIN)

__global__ void k_init(unsigned* amax) {
  if (blockIdx.x == 0 && threadIdx.x == 0) amax[0] = 0u;
}

__global__ void k_zero(int4* __restrict__ p, int n16) {
  int4 z = {0, 0, 0, 0};
  for (int i = blockIdx.x * blockDim.x + threadIdx.x; i < n16;
       i += gridDim.x * blockDim.x)
    p[i] = z;
}

__global__ void k_absmax(const float* __restrict__ x, unsigned* amax, int n4) {
  const float4* x4 = (const float4*)x;
  float m = 0.f;
  for (int i = blockIdx.x * blockDim.x + threadIdx.x; i < n4;
       i += gridDim.x * blockDim.x) {
    float4 v = x4[i];
    m = fmaxf(m, fmaxf(fmaxf(fabsf(v.x), fabsf(v.y)),
                       fmaxf(fabsf(v.z), fabsf(v.w))));
  }
  for (int o = 16; o > 0; o >>= 1) m = fmaxf(m, __shfl_xor(m, o, 32));
  __shared__ float sm[8];
  int wave = threadIdx.x >> 5, lane = threadIdx.x & 31;
  if (lane == 0) sm[wave] = m;
  __syncthreads();
  if (threadIdx.x == 0) {
    float mm = sm[0];
    for (int i = 1; i < 8; ++i) mm = fmaxf(mm, sm[i]);
    atomicMax(amax, __float_as_uint(mm));
  }
}

__global__ void k_qweight(const float* __restrict__ weight,
                          const float* __restrict__ bias,
                          const unsigned* __restrict__ amax,
                          const int* __restrict__ hs_p,
                          double* __restrict__ fq, int* __restrict__ bq,
                          signed char* __restrict__ wq) {
  int co  = blockIdx.x;
  int tid = threadIdx.x;
  __shared__ float red[256];
  __shared__ float s_sw;
  float m = 0.f;
  for (int i = tid; i < NTAPS * CIN; i += 256) {
    int t = i / CIN, c = i - t * CIN;
    int ky = (t < 10) ? (t / 5) : 2;
    int kx = (t < 10) ? (t % 5) : (t - 10);
    float wv = weight[(((size_t)co * CIN + c) * KW + ky) * KW + kx];
    m = fmaxf(m, fabsf(wv));
  }
  red[tid] = m;
  __syncthreads();
  for (int s = 128; s > 0; s >>= 1) {
    if (tid < s) red[tid] = fmaxf(red[tid], red[tid + s]);
    __syncthreads();
  }
  if (tid == 0) {
    float s_a = fmaxf(__uint_as_float(amax[0]) / 127.0f, 1e-8f);
    float s_w = fmaxf(red[0] / 127.0f, 1e-8f);
    s_sw = s_w;
    float scale = s_a * s_w;
    int er;
    double mant  = frexp((double)scale, &er);
    double m_int = floor(mant * 2147483648.0 + 0.5);
    int e  = 31 - er;
    int hs = hs_p[0];
    fq[co] = ldexp(m_int, hs - e);
    bq[co] = (int)rintf(bias[co] / scale);
  }
  __syncthreads();
  float s_w = s_sw;
  for (int i = tid; i < NTAPS * CIN; i += 256) {
    int t = i / CIN, c = i - t * CIN;
    int ky = (t < 10) ? (t / 5) : 2;
    int kx = (t < 10) ? (t % 5) : (t - 10);
    float wv = weight[(((size_t)co * CIN + c) * KW + ky) * KW + kx];
    float q  = fminf(fmaxf(rintf(wv / s_w), -128.f), 127.f);
    wq[(size_t)co * KGEMM + t * CIN + c] = (signed char)q;
  }
}

__global__ void k_qx(const float* __restrict__ x,
                     const unsigned* __restrict__ amax,
                     signed char* __restrict__ xq) {
  int bh = blockIdx.x;
  int b  = bh >> 6, h = bh & 63;
  float s_a = fmaxf(__uint_as_float(amax[0]) / 127.0f, 1e-8f);
  __shared__ __align__(16) signed char lds[W_ * CIN];
  for (int i = threadIdx.x; i < W_ * CIN; i += blockDim.x) {
    int w = i & 63, c = i >> 6;
    float v = x[(((size_t)b * CIN + c) * H_ + h) * W_ + w];
    float q = fminf(fmaxf(rintf(v / s_a), -128.f), 127.f);
    lds[w * CIN + c] = (signed char)q;
  }
  __syncthreads();
  const int* ldsi = (const int*)lds;
  int* dst = (int*)(xq + ((size_t)((b * HP + h + 2) * WP + 2)) * CIN);
  for (int i = threadIdx.x; i < (W_ * CIN) / 4; i += blockDim.x) dst[i] = ldsi[i];
}

union Frag { long long l[4]; v8i v; };

__device__ __forceinline__ void load_A(Frag& A, const signed char* ap) {
#pragma unroll
  for (int j = 0; j < 4; ++j)
    A.l[j] = *(const long long*)(ap + (j & 1) * 8 + (j >> 1) * 32);
}
__device__ __forceinline__ void load_B(Frag& Bv, const signed char* bp) {
  Bv.l[0] = ((const long long*)bp)[0];
  Bv.l[1] = ((const long long*)bp)[1];
  Bv.l[2] = ((const long long*)bp)[4];
  Bv.l[3] = ((const long long*)bp)[5];
}
__device__ __forceinline__ constexpr int a_off(int it) {
  int t = it / 3, kc = it - 3 * t;
  int dy = (t < 10) ? (t / 5 - 2) : 0;
  int dx = (t < 10) ? (t % 5 - 2) : (t - 12);
  return (dy * WP + dx) * CIN + kc * 64;
}

__global__ __launch_bounds__(256) void k_conv(
    const signed char* __restrict__ xq, const signed char* __restrict__ wq,
    const double* __restrict__ fq, const int* __restrict__ bq,
    float* __restrict__ out) {
  int lane  = threadIdx.x & 31;
  int wvid  = threadIdx.x >> 5;
  int wid   = blockIdx.x * 8 + wvid;
  int strip = wid / 6;
  int ng    = wid - strip * 6;
  int lm    = lane & 15;
  int hsel  = lane >> 4;
  int p  = strip * 16 + lm;
  int pb = p >> 12;
  int ph = (p >> 6) & 63;
  int pw = p & 63;
  int kbA = hsel * 16;
  int kbB = hsel * 16;

  const signed char* ap0 =
      xq + ((size_t)((pb * HP + ph + 2) * WP + pw + 2)) * CIN + kbA;

  const signed char* bbase[4];
#pragma unroll
  for (int tt = 0; tt < 4; ++tt)
    bbase[tt] = wq + (size_t)(ng * 64 + tt * 16 + lm) * KGEMM + kbB;

  v8i acc[4];
#pragma unroll
  for (int tt = 0; tt < 4; ++tt) acc[tt] = (v8i){0, 0, 0, 0, 0, 0, 0, 0};

  Frag A[2], Bf[2][4];
  load_A(A[0], ap0 + a_off(0));
#pragma unroll
  for (int tt = 0; tt < 4; ++tt) load_B(Bf[0][tt], bbase[tt]);

#pragma unroll
  for (int it = 0; it < NITER; ++it) {
    const int cur = it & 1, nxt = cur ^ 1;
    if (it + 1 < NITER) {
      load_A(A[nxt], ap0 + a_off(it + 1));
#pragma unroll
      for (int tt = 0; tt < 4; ++tt)
        load_B(Bf[nxt][tt], bbase[tt] + (it + 1) * 64);
    }
#pragma unroll
    for (int tt = 0; tt < 4; ++tt)
      acc[tt] = __builtin_amdgcn_wmma_i32_16x16x64_iu8(
          true, A[cur].v, true, Bf[cur][tt].v, acc[tt], false, false);
  }

  __shared__ int tlds[8][256];
  int* myl = tlds[wvid];
  int pr = strip * 16 + hsel * 8;
  int ob = pr >> 12, oh = (pr >> 6) & 63, ow = pr & 63;
#pragma unroll
  for (int tt = 0; tt < 4; ++tt) {
#pragma unroll
    for (int r = 0; r < 8; ++r)
      myl[(r + hsel * 8) * 16 + lm] = acc[tt][r];
    __builtin_amdgcn_wave_barrier();
    int co   = ng * 64 + tt * 16 + lm;
    double f = fq[co];
    int bi   = bq[co];
    float ov[8];
#pragma unroll
    for (int j = 0; j < 8; ++j) {
      int v = myl[(hsel * 8 + j) * 16 + lm] + bi;
      ov[j] = (float)rint((double)v * f);
    }
    float* op = out + (((size_t)ob * COUT + co) * H_ + oh) * W_ + ow;
    *(float4*)(op)     = make_float4(ov[0], ov[1], ov[2], ov[3]);
    *(float4*)(op + 4) = make_float4(ov[4], ov[5], ov[6], ov[7]);
    __builtin_amdgcn_wave_barrier();
  }
}

extern "C" void kernel_launch(void* const* d_in, const int* in_sizes, int n_in,
                              void* d_out, int out_size, void* d_ws,
                              size_t ws_size, hipStream_t stream) {
  const float* x      = (const float*)d_in[0];
  const float* weight = (const float*)d_in[1];
  const float* bias   = (const float*)d_in[2];
  const int* hs       = (const int*)d_in[4];

  unsigned*    amax = (unsigned*)d_ws;
  double*      fq   = (double*)((char*)d_ws + WS_F_OFF);
  int*         bq   = (int*)((char*)d_ws + WS_BQ_OFF);
  signed char* wq   = (signed char*)d_ws + WS_WQ_OFF;
  signed char* xq   = (signed char*)d_ws + WS_XQ_OFF;
  float*       out  = (float*)d_out;

  k_init<<<1, 64, 0, stream>>>(amax);
  k_zero<<<512, 256, 0, stream>>>((int4*)xq, XQ_BYTES / 16);
  k_absmax<<<768, 256, 0, stream>>>(x, amax, (B_ * CIN * H_ * W_) / 4);
  k_qweight<<<COUT, 256, 0, stream>>>(weight, bias, amax, hs, fq, bq, wq);
  k_qx<<<B_ * H_, 256, 0, stream>>>(x, amax, xq);
  k_conv<<<1536, 256, 0, stream>>>(xq, wq, fq, bq, out);
}
